// GraphConv_16449724744756
// MI455X (gfx1250) — matrix operand order repaired and checked by reading, not compiled, not run on hardware
//
#include <hip/hip_runtime.h>
#include <hip/hip_bf16.h>
#include <stdint.h>

// ---------------------------------------------------------------------------
// MI455X (gfx1250) fused GraphConv. wave32, WMMA f32_16x16x32_f16.
// Key idea: never materialize the E x 128 x 128 per-edge weight tensor (1 GB).
// BN of layer 4 is computed analytically from the Gram matrix of x3, so the
// layer-4 GEMM fuses with the einsum: y-tiles stay in WMMA accumulators.
// New this round: x_i staging uses gfx1250 async global->LDS gather
// (global_load_async_to_lds_b32 + s_wait_asynccnt) doing the transpose
// scatter directly into LDS with no VGPR round-trip.
// ---------------------------------------------------------------------------

typedef __attribute__((ext_vector_type(16))) _Float16 v16h;
typedef __attribute__((ext_vector_type(8)))  float    v8f;
typedef __attribute__((ext_vector_type(4)))  unsigned v4u;

union F16Frag { v16h v; v4u q[2]; };

#define BN_EPS 1e-5f

// A-fragment (16x32 f16, M x K): lane l holds row M=l&15.
// halves 0..7  -> K = base + (l<16?0:8)  + 0..7   (contiguous 16B)
// halves 8..15 -> K = base + (l<16?0:8) + 16..23  (contiguous 16B)
__device__ __forceinline__ v16h load_a_frag(const _Float16* __restrict__ A,
                                            int lda, int row, int kbase, int lane) {
  const _Float16* p = A + (size_t)row * lda + kbase + ((lane & 16) ? 8 : 0);
  F16Frag f;
  f.q[0] = *reinterpret_cast<const v4u*>(p);
  f.q[1] = *reinterpret_cast<const v4u*>(p + 16);
  return f.v;
}

// B-fragment (32x16 f16, K x N): lane l holds col N=l&15.
// halves 0..7  -> K = base + (l<16?0:8)  + 0..7   (contiguous 16B)
// halves 8..15 -> K = base + (l<16?0:8) + 16..23  (contiguous 16B)
__device__ __forceinline__ v16h load_b_frag(const _Float16* __restrict__ W,
                                            int ldw, int nrow, int kbase, int lane) {
  const _Float16* p = W + (size_t)nrow * ldw + kbase + ((lane & 16) ? 8 : 0);
  F16Frag f;
  f.q[0] = *reinterpret_cast<const v4u*>(p);
  f.q[1] = *reinterpret_cast<const v4u*>(p + 16);
  return f.v;
}

// ---------------------------------------------------------------------------
// f32 -> f16 convert with optional K padding (zero fill).
__global__ void cvt_pad_h(const float* __restrict__ src, _Float16* __restrict__ dst,
                          int rows, int K, int Kp) {
  int i = blockIdx.x * 256 + threadIdx.x;
  int total = rows * Kp;
  if (i >= total) return;
  int r = i / Kp, k = i - r * Kp;
  dst[i] = (k < K) ? (_Float16)src[(size_t)r * K + k] : (_Float16)0.0f;
}

__global__ void zerof(float* __restrict__ p, int n) {
  int i = blockIdx.x * 256 + threadIdx.x;
  if (i < n) p[i] = 0.0f;
}

// ---------------------------------------------------------------------------
// C[M x N] = A[M x K](f16) @ W[N x K](f16)^T, fp32 accumulate via WMMA.
// block = 256 (8 waves), wave w covers cols [bY*128 + 16w, +16), rows = 16*bX.
__global__ __launch_bounds__(256) void wmma_gemm(const _Float16* __restrict__ A,
                                                 const _Float16* __restrict__ W,
                                                 float* __restrict__ C,
                                                 int N, int K) {
  int lane = threadIdx.x & 31;
  int wave = threadIdx.x >> 5;
  int row0 = blockIdx.x * 16;
  int n0   = blockIdx.y * 128 + wave * 16;
  int arow = row0 + (lane & 15);
  int brow = n0 + (lane & 15);
  v8f acc = {};
  for (int kq = 0; kq < K; kq += 32) {
    v16h a = load_a_frag(A, K, arow, kq, lane);
    v16h b = load_b_frag(W, K, brow, kq, lane);
    acc = __builtin_amdgcn_wmma_f32_16x16x32_f16(false, a, false, b,
                                                 (short)0, acc, false, false);
  }
  int col   = n0 + (lane & 15);
  int rbase = row0 + ((lane & 16) ? 8 : 0);
#pragma unroll
  for (int j = 0; j < 8; ++j)
    C[(size_t)(rbase + j) * N + col] = acc[j];
}

// ---------------------------------------------------------------------------
// Per-column mean / inv-std over E rows (one block per column, deterministic).
__global__ __launch_bounds__(256) void colstats(const float* __restrict__ z, int E, int N,
                                                float* __restrict__ mean,
                                                float* __restrict__ invstd) {
  int c = blockIdx.x, tid = threadIdx.x;
  float s = 0.f, s2 = 0.f;
  for (int r = tid; r < E; r += 256) {
    float v = z[(size_t)r * N + c];
    s += v; s2 += v * v;
  }
  __shared__ float r1[256], r2[256];
  r1[tid] = s; r2[tid] = s2;
  __syncthreads();
  for (int d = 128; d > 0; d >>= 1) {
    if (tid < d) { r1[tid] += r1[tid + d]; r2[tid] += r2[tid + d]; }
    __syncthreads();
  }
  if (tid == 0) {
    float m = r1[0] / (float)E;
    float v = r2[0] / (float)E - m * m;
    mean[c] = m;
    invstd[c] = rsqrtf(v + BN_EPS);
  }
}

// BN + ReLU, writes f16 (next GEMM operand) and optionally f32.
__global__ void bn_relu(const float* __restrict__ z, const float* __restrict__ mean,
                        const float* __restrict__ invstd, const float* __restrict__ g,
                        const float* __restrict__ b, float* __restrict__ xf,
                        _Float16* __restrict__ xh, int total, int N) {
  int i = blockIdx.x * 256 + threadIdx.x;
  if (i >= total) return;
  int c = i % N;
  float v = (z[i] - mean[c]) * invstd[c] * g[c] + b[c];
  v = v > 0.f ? v : 0.f;
  if (xf) xf[i] = v;
  xh[i] = (_Float16)v;
}

// ---------------------------------------------------------------------------
// S = x3^T x3 / E  (128x128 Gram), LDS-tiled over the E dimension.
__global__ __launch_bounds__(256) void gram128(const float* __restrict__ x3, int E,
                                               float* __restrict__ S) {
  __shared__ float As[64][16], Bs[64][16];
  int tx = threadIdx.x, ty = threadIdx.y;
  int tid = ty * 16 + tx;
  float s = 0.f;
  for (int e0 = 0; e0 < E; e0 += 64) {
    for (int t = tid; t < 64 * 16; t += 256) {
      int r = t >> 4, c = t & 15;
      As[r][c] = x3[(size_t)(e0 + r) * 128 + blockIdx.y * 16 + c];
      Bs[r][c] = x3[(size_t)(e0 + r) * 128 + blockIdx.x * 16 + c];
    }
    __syncthreads();
#pragma unroll 8
    for (int r = 0; r < 64; ++r) s += As[r][ty] * Bs[r][tx];
    __syncthreads();
  }
  int i = blockIdx.y * 16 + ty, j = blockIdx.x * 16 + tx;
  S[i * 128 + j] = s / (float)E;
}

// Analytic BN-4 fold: for channel c, mean_c = W4_c . mu3,
// E[y^2]_c = W4_c^T S W4_c, var = E[y^2]-mean^2; st4[c] = (g*rstd, b - mean*g*rstd).
__global__ __launch_bounds__(128) void prep_st4(const float* __restrict__ W4,
                                                const float* __restrict__ S,
                                                const float* __restrict__ mu3,
                                                const float* __restrict__ g4,
                                                const float* __restrict__ b4,
                                                float2* __restrict__ st4) {
  int c = blockIdx.x, r = threadIdx.x;
  __shared__ float wc[128], r1[128], r2[128];
  wc[r] = W4[(size_t)c * 128 + r];
  __syncthreads();
  float q = 0.f;
  const float* Sr = S + r * 128;
#pragma unroll 8
  for (int j = 0; j < 128; ++j) q = fmaf(Sr[j], wc[j], q);
  r1[r] = wc[r] * mu3[r];
  r2[r] = wc[r] * q;
  __syncthreads();
  for (int d = 64; d > 0; d >>= 1) {
    if (r < d) { r1[r] += r1[r + d]; r2[r] += r2[r + d]; }
    __syncthreads();
  }
  if (r == 0) {
    float mean = r1[0];
    float var  = r2[0] - mean * mean;
    float sc   = g4[c] * rsqrtf(var + BN_EPS);
    st4[c] = make_float2(sc, b4[c] - mean * sc);
  }
}

// ---------------------------------------------------------------------------
// Fused layer-4 GEMM + BN fold + einsum + segment-sum scatter.
// 32 edges per block (2 M-tiles), 8 waves: wave w owns k-cols [16w, 16w+16).
// For each d in 0..127: y(16x16) = x3_tile @ W4[d*128+k, :]^T via 4 chained
// WMMAs; msg += x_i[:,d] * (y*s + t). 8192 WMMAs per block (1024 per wave), W4 streamed from L2.
// x_i is gathered + transposed straight into LDS with async global->LDS copies
// (per-lane global address AND per-lane LDS address), tracked by ASYNCcnt.
__global__ __launch_bounds__(256) void fused_msg(const _Float16* __restrict__ x3h,
                                                 const _Float16* __restrict__ W4h,
                                                 const float2* __restrict__ st4,
                                                 const float* __restrict__ node,
                                                 const int* __restrict__ eidx,
                                                 float* __restrict__ sums,
                                                 float* __restrict__ cnt) {
  __shared__ float xiT[128][32];   // x_i transposed: [d][edge_local]
  __shared__ int dstS[32];
  int tid = threadIdx.x, lane = tid & 31, wave = tid >> 5;
  int e0 = blockIdx.x * 32;

  // async gather-transpose: node[src[el]][d] -> LDS xiT[d][el]
  for (int idx = tid; idx < 32 * 128; idx += 256) {
    int el = idx >> 7, d = idx & 127;
    int src = eidx[(size_t)(e0 + el) * 2 + 0];
    const float* gp = node + (size_t)src * 128 + d;
    unsigned ldsoff = (unsigned)(size_t)&xiT[d][el];
    asm volatile("global_load_async_to_lds_b32 %0, %1, off"
                 :: "v"(ldsoff), "v"(gp)
                 : "memory");
  }
  if (tid < 32) dstS[tid] = eidx[(size_t)(e0 + tid) * 2 + 1];
  asm volatile("s_wait_asynccnt 0x0" ::: "memory");  // own wave's asyncs done
  __syncthreads();                                   // all waves' LDS visible

  int k0  = wave * 16;
  int col = k0 + (lane & 15);
  int off = (lane & 16) ? 8 : 0;

  // A fragments: held in registers for the whole tile (2 M-tiles x 4 K-chunks).
  v16h Af[2][4];
#pragma unroll
  for (int m = 0; m < 2; ++m)
#pragma unroll
    for (int kc = 0; kc < 4; ++kc)
      Af[m][kc] = load_a_frag(x3h, 128, e0 + m * 16 + (lane & 15), kc * 32, lane);

  v8f acc0 = {}, acc1 = {};

  for (int d = 0; d < 128; ++d) {
    const _Float16* bp = W4h + (size_t)(d * 128 + col) * 128;
    __builtin_prefetch(bp + 128 * 128, 0, 1);   // next d row block (speculative)
    v16h B0 = load_b_frag(W4h, 128, d * 128 + col, 0,  lane);
    v16h B1 = load_b_frag(W4h, 128, d * 128 + col, 32, lane);
    v16h B2 = load_b_frag(W4h, 128, d * 128 + col, 64, lane);
    v16h B3 = load_b_frag(W4h, 128, d * 128 + col, 96, lane);
    float2 st = st4[d * 128 + col];

#pragma unroll
    for (int m = 0; m < 2; ++m) {
      v8f y = {};
      y = __builtin_amdgcn_wmma_f32_16x16x32_f16(false, Af[m][0], false, B0, (short)0, y, false, false);
      y = __builtin_amdgcn_wmma_f32_16x16x32_f16(false, Af[m][1], false, B1, (short)0, y, false, false);
      y = __builtin_amdgcn_wmma_f32_16x16x32_f16(false, Af[m][2], false, B2, (short)0, y, false, false);
      y = __builtin_amdgcn_wmma_f32_16x16x32_f16(false, Af[m][3], false, B3, (short)0, y, false, false);

      const float* xp = &xiT[d][m * 16 + off];  // broadcast across half-wave
      float4 xa = *reinterpret_cast<const float4*>(xp);
      float4 xb = *reinterpret_cast<const float4*>(xp + 4);
      v8f& a = (m == 0) ? acc0 : acc1;
      a[0] = fmaf(xa.x, fmaf(y[0], st.x, st.y), a[0]);
      a[1] = fmaf(xa.y, fmaf(y[1], st.x, st.y), a[1]);
      a[2] = fmaf(xa.z, fmaf(y[2], st.x, st.y), a[2]);
      a[3] = fmaf(xa.w, fmaf(y[3], st.x, st.y), a[3]);
      a[4] = fmaf(xb.x, fmaf(y[4], st.x, st.y), a[4]);
      a[5] = fmaf(xb.y, fmaf(y[5], st.x, st.y), a[5]);
      a[6] = fmaf(xb.z, fmaf(y[6], st.x, st.y), a[6]);
      a[7] = fmaf(xb.w, fmaf(y[7], st.x, st.y), a[7]);
    }
  }

  // scatter into segment sums (hardware global_atomic_add_f32)
#pragma unroll
  for (int m = 0; m < 2; ++m) {
    const v8f& a = (m == 0) ? acc0 : acc1;
    int elb = m * 16 + off;
#pragma unroll
    for (int j = 0; j < 8; ++j) {
      int dst = dstS[elb + j];
      unsafeAtomicAdd(&sums[(size_t)dst * 128 + col], a[j]);
    }
  }
  if (tid < 32) unsafeAtomicAdd(&cnt[dstS[tid]], 1.0f);
}

// ---------------------------------------------------------------------------
// segment mean -> relu(+bias) -> GRU cell. One block (128 thr) per node.
__global__ __launch_bounds__(128) void gru_out(const float* __restrict__ sums,
                                               const float* __restrict__ cnt,
                                               const float* __restrict__ bias,
                                               const float* __restrict__ h0,
                                               const float* __restrict__ W_ih,
                                               const float* __restrict__ W_hh,
                                               const float* __restrict__ b_ih,
                                               const float* __restrict__ b_hh,
                                               float* __restrict__ out, int N) {
  int n = blockIdx.x, tid = threadIdx.x;
  __shared__ float ms[128], hs[128], gis[384], ghs[384];
  float c = cnt[n]; c = c > 1.f ? c : 1.f;
  float m = sums[(size_t)n * 128 + tid] / c + bias[tid];
  ms[tid] = m > 0.f ? m : 0.f;
  hs[tid] = h0[(size_t)n * 128 + tid];
  __syncthreads();
  for (int o = tid; o < 384; o += 128) {
    const float* wi = W_ih + (size_t)o * 128;
    const float* wh = W_hh + (size_t)o * 128;
    float a = b_ih[o], bb = b_hh[o];
#pragma unroll 8
    for (int k = 0; k < 128; ++k) {
      a  = fmaf(wi[k], ms[k], a);
      bb = fmaf(wh[k], hs[k], bb);
    }
    gis[o] = a; ghs[o] = bb;
  }
  __syncthreads();
  float r  = 1.f / (1.f + expf(-(gis[tid] + ghs[tid])));
  float zz = 1.f / (1.f + expf(-(gis[128 + tid] + ghs[128 + tid])));
  float nn = tanhf(gis[256 + tid] + r * ghs[256 + tid]);
  float hn = (1.f - zz) * nn + zz * hs[tid];
  out[(size_t)n * 128 + tid] = hn;
  out[(size_t)N * 128 + (size_t)n * 128 + tid] = hn;
}

// ---------------------------------------------------------------------------
extern "C" void kernel_launch(void* const* d_in, const int* in_sizes, int n_in,
                              void* d_out, int out_size, void* d_ws, size_t ws_size,
                              hipStream_t stream) {
  const float* node   = (const float*)d_in[0];
  const int*   eidx   = (const int*)  d_in[1];
  const float* edge   = (const float*)d_in[2];
  const float* hidden = (const float*)d_in[3];
  const float* W1 = (const float*)d_in[4];
  const float* g1 = (const float*)d_in[5];
  const float* b1 = (const float*)d_in[6];
  const float* W2 = (const float*)d_in[7];
  const float* g2 = (const float*)d_in[8];
  const float* b2 = (const float*)d_in[9];
  const float* W3 = (const float*)d_in[10];
  const float* g3 = (const float*)d_in[11];
  const float* b3 = (const float*)d_in[12];
  const float* W4 = (const float*)d_in[13];
  const float* g4 = (const float*)d_in[14];
  const float* b4 = (const float*)d_in[15];
  const float* bias = (const float*)d_in[16];
  const float* W_ih = (const float*)d_in[17];
  const float* W_hh = (const float*)d_in[18];
  const float* b_ih = (const float*)d_in[19];
  const float* b_hh = (const float*)d_in[20];
  float* out = (float*)d_out;

  const int N  = in_sizes[0] / 128;      // 4096
  const int E  = in_sizes[2] / 16;       // 16384
  const int R4 = in_sizes[13] / 128;     // 16384 = D*D

  // workspace carve-up (256B aligned)
  char* ws = (char*)d_ws;
  size_t off = 0;
  auto carve = [&](size_t bytes) -> void* {
    void* p = ws + off;
    off = (off + bytes + 255) & ~(size_t)255;
    return p;
  };
  _Float16* edge_h = (_Float16*)carve((size_t)E * 32 * 2);
  _Float16* W1h    = (_Float16*)carve((size_t)256 * 32 * 2);
  _Float16* W2h    = (_Float16*)carve((size_t)256 * 256 * 2);
  _Float16* W3h    = (_Float16*)carve((size_t)128 * 256 * 2);
  _Float16* W4h    = (_Float16*)carve((size_t)R4 * 128 * 2);
  float*    zbuf   = (float*)   carve((size_t)E * 256 * 4);   // reused by L1 & L2
  _Float16* x1h    = (_Float16*)carve((size_t)E * 256 * 2);
  _Float16* x2h    = (_Float16*)carve((size_t)E * 256 * 2);
  float*    z3     = (float*)   carve((size_t)E * 128 * 4);
  _Float16* x3h    = (_Float16*)carve((size_t)E * 128 * 2);
  float*    x3f    = (float*)   carve((size_t)E * 128 * 4);
  float*    meanb  = (float*)   carve(256 * 4);
  float*    invb   = (float*)   carve(256 * 4);
  float*    mu3    = (float*)   carve(128 * 4);
  float*    Smat   = (float*)   carve((size_t)128 * 128 * 4);
  float2*   st4    = (float2*)  carve((size_t)R4 * 8);
  float*    sums   = (float*)   carve((size_t)N * 128 * 4);
  float*    cntb   = (float*)   carve((size_t)N * 4);

  auto grid1 = [](int total) { return dim3((total + 255) / 256); };

  // f32 -> f16 conversions (K padded to 32 for layer 1)
  cvt_pad_h<<<grid1(E * 32),      256, 0, stream>>>(edge, edge_h, E, 16, 32);
  cvt_pad_h<<<grid1(256 * 32),    256, 0, stream>>>(W1, W1h, 256, 16, 32);
  cvt_pad_h<<<grid1(256 * 256),   256, 0, stream>>>(W2, W2h, 256, 256, 256);
  cvt_pad_h<<<grid1(128 * 256),   256, 0, stream>>>(W3, W3h, 128, 256, 256);
  cvt_pad_h<<<grid1(R4 * 128),    256, 0, stream>>>(W4, W4h, R4, 128, 128);

  // layer 1: z = edge @ W1^T ; BN ; ReLU
  wmma_gemm<<<dim3(E / 16, 2), 256, 0, stream>>>(edge_h, W1h, zbuf, 256, 32);
  colstats <<<256, 256, 0, stream>>>(zbuf, E, 256, meanb, invb);
  bn_relu  <<<grid1(E * 256), 256, 0, stream>>>(zbuf, meanb, invb, g1, b1, nullptr, x1h, E * 256, 256);

  // layer 2
  wmma_gemm<<<dim3(E / 16, 2), 256, 0, stream>>>(x1h, W2h, zbuf, 256, 256);
  colstats <<<256, 256, 0, stream>>>(zbuf, E, 256, meanb, invb);
  bn_relu  <<<grid1(E * 256), 256, 0, stream>>>(zbuf, meanb, invb, g2, b2, nullptr, x2h, E * 256, 256);

  // layer 3 (keep f32 copy for the analytic BN-4 stats)
  wmma_gemm<<<dim3(E / 16, 1), 256, 0, stream>>>(x2h, W3h, z3, 128, 256);
  colstats <<<128, 256, 0, stream>>>(z3, E, 128, meanb, invb);
  bn_relu  <<<grid1(E * 128), 256, 0, stream>>>(z3, meanb, invb, g3, b3, x3f, x3h, E * 128, 128);

  // analytic BN-4: mu3, Gram, per-channel (scale, shift)
  colstats<<<128, 256, 0, stream>>>(x3f, E, 128, mu3, invb /*unused*/);
  gram128 <<<dim3(8, 8), dim3(16, 16), 0, stream>>>(x3f, E, Smat);
  prep_st4<<<R4, 128, 0, stream>>>(W4, Smat, mu3, g4, b4, st4);

  // fused GEMM4 + einsum + segment scatter
  zerof<<<grid1(N * 128), 256, 0, stream>>>(sums, N * 128);
  zerof<<<grid1(N),       256, 0, stream>>>(cntb, N);
  fused_msg<<<E / 32, 256, 0, stream>>>(x3h, W4h, st4, node, eidx, sums, cntb);

  // segment mean + bias + relu + GRU, writes (hn, hn[None]) concatenated
  gru_out<<<N, 128, 0, stream>>>(sums, cntb, bias, hidden, W_ih, W_hh, b_ih, b_hh, out, N);
}
